// StackedQLSTM_28716151341215
// MI455X (gfx1250) — matrix operand order repaired and checked by reading, not compiled, not run on hardware
//
#include <hip/hip_runtime.h>

// StackedQLSTM for MI455X (gfx1250, wave32, WMMA).
// Batch-parallel: 8 WGs x 16 rows, zero inter-WG sync. Per step/WG:
//   [16 x 288] x [288 x (4*256+16)] bf16 WMMA GEMM, A = [h_{t-1} | x_t | 1 | 0],
//   B rows 256..260 carry the input projection Wx and biases (no VALU projection).
// c state in VGPRs, h in LDS ping-pong; weights pre-swizzled bf16 in L2.
// Anti-hoist: only a per-lane 32-bit offset is made opaque per step, so loads
// stay global_load_b128 (saddr + voffset + imm) and cannot be hoisted/spilled.

#define TT 512
#define BBATCH 128

typedef __attribute__((ext_vector_type(16))) __bf16       v16bf;
typedef __attribute__((ext_vector_type(8)))  float        v8f;
typedef __attribute__((ext_vector_type(4)))  unsigned int u32x4;
typedef __attribute__((ext_vector_type(4)))  float        f32x4;

union FragBF { u32x4 u[2]; v16bf v; };

// quaternion 4x4 block structure (row block a, col block b): component idx + sign
__constant__ int   qidx[4][4] = {{0,1,2,3},{1,0,3,2},{2,3,0,1},{3,2,1,0}};
__constant__ float qsgn[4][4] = {{ 1, 1, 1, 1},{-1, 1, 1,-1},{-1,-1, 1, 1},{-1, 1,-1, 1}};

// raw-rate activations: v_exp_f32 / v_rcp_f32 are TRANS ops -> co-execute with WMMA
__device__ __forceinline__ float sigm(float x) {
    return __builtin_amdgcn_rcpf(1.0f + __expf(-x));
}
__device__ __forceinline__ float tanh_fast(float x) {
    return 2.0f * __builtin_amdgcn_rcpf(1.0f + __expf(-2.0f * x)) - 1.0f;
}

// ---------------------------------------------------------------------------
// Prep: build augmented, pre-swizzled bf16 B fragments.
// Ubs[l][nt(64)][kc(9)][lane(32)][i(16)]; for element i of lane:
//   K = kc*32 + (i%8) + (i/8)*16 + (lane/16)*8   (32x16 bf16 B-fragment layout)
//   N = nt*16 + lane%16 ; gate g = nt/16 ; col = (nt%16)*16 + lane%16
// Rows K<256: quaternion-expanded Uh. K=256..259: quaternion-expanded Wx.
// K=260: gate bias wb. K>260: 0.
// ---------------------------------------------------------------------------
__global__ void prep_ubs(const float* __restrict__ ur, const float* __restrict__ ui,
                         const float* __restrict__ uj, const float* __restrict__ uk,
                         const float* __restrict__ wr, const float* __restrict__ wi,
                         const float* __restrict__ wj, const float* __restrict__ wk,
                         const float* __restrict__ wb, __bf16* __restrict__ ubs) {
    int e = blockIdx.x * 256 + threadIdx.x;
    if (e >= 2 * 64 * 9 * 32 * 16) return;
    int i    = e & 15;
    int lane = (e >> 4) & 31;
    int t2   = e >> 9;
    int kc   = t2 % 9;
    int t3   = t2 / 9;
    int nt   = t3 & 63;
    int l    = t3 >> 6;
    int K    = kc * 32 + (i & 7) + ((i >> 3) << 4) + ((lane >> 4) << 3);
    int g    = nt >> 4;
    int kcol = ((nt & 15) << 4) + (lane & 15);
    int b = kcol >> 6, cc = kcol & 63;
    const float* ucomp[4] = {ur, ui, uj, uk};
    const float* wcomp[4] = {wr, wi, wj, wk};
    float v;
    if (K < 256) {
        int a = K >> 6, rr = K & 63;
        v = qsgn[a][b] * ucomp[qidx[a][b]][((l * 4 + g) * 64 + rr) * 64 + cc];
    } else if (K < 260) {
        int f = K - 256;                       // Wx row block (F//4==1 -> rr==0)
        v = qsgn[f][b] * wcomp[qidx[f][b]][(l * 4 + g) * 64 + cc];
    } else if (K == 260) {
        v = wb[(l * 4 + g) * 256 + kcol];      // bias row (A carries constant 1)
    } else {
        v = 0.0f;
    }
    ubs[e] = (__bf16)v;
}

// fco tile: fws[l][kc(9)][lane(32)][i(16)]; cols f<4: K<256 -> fco_w[l][K][f],
// K==260 -> fco_b[l][f]; everything else 0.
__global__ void prep_fws(const float* __restrict__ fw, const float* __restrict__ fb,
                         __bf16* __restrict__ fws) {
    int e = blockIdx.x * 256 + threadIdx.x;
    if (e >= 2 * 9 * 32 * 16) return;
    int i    = e & 15;
    int lane = (e >> 4) & 31;
    int kc   = (e >> 9) % 9;
    int l    = (e >> 9) / 9;
    int K = kc * 32 + (i & 7) + ((i >> 3) << 4) + ((lane >> 4) << 3);
    int f = lane & 15;
    float v = 0.0f;
    if (f < 4) {
        if (K < 256)       v = fw[(l * 256 + K) * 4 + f];
        else if (K == 260) v = fb[l * 4 + f];
    }
    fws[e] = (__bf16)v;
}

// ---------------------------------------------------------------------------
// fco finalize: b = bb + r + 8*(lane/16), f = lane%16 (valid f<4); bias already
// in accumulator via the constant-1 A column. L2-normalize over f (shfl quad).
// ---------------------------------------------------------------------------
__device__ __forceinline__ void fco_store(v8f afc, float* __restrict__ dst,
                                          int bb, int lhi, int ln) {
#pragma unroll
    for (int r = 0; r < 8; ++r) {
        float o = afc[r];
        float s = o * o;
        s += __shfl_xor(s, 1, 32);
        s += __shfl_xor(s, 2, 32);
        float nrm = fmaxf(__builtin_amdgcn_sqrtf(s), 1e-12f);
        if (ln < 4) dst[(bb + r + (lhi << 3)) * 4 + ln] = o * __builtin_amdgcn_rcpf(nrm);
    }
}

// ---------------------------------------------------------------------------
// Main persistent scan: 8 blocks x 512 threads (16 waves).
// Wave w owns hidden slice k = w*16 + lane%16 and gate tiles nt = g*16 + w.
// ---------------------------------------------------------------------------
__global__ __launch_bounds__(512, 1) void qlstm_main(
    const float* __restrict__ x, const __bf16* __restrict__ Ubs,
    const __bf16* __restrict__ fws, float* __restrict__ x1, float* __restrict__ out) {

    // [16 rows][296 bf16]: cols 0..255 = h, 256..259 = x_t, 260 = 1, 261.. = 0/pad.
    // Row stride 592B = 148 DW -> 16 A-load lanes hit distinct banks.
    __shared__ __align__(16) __bf16 hbuf[2][16][296];

    const int tid  = threadIdx.x;
    const int w    = tid >> 5;
    const int lane = tid & 31;
    const int lhi  = lane >> 4;
    const int ln   = lane & 15;
    const int k    = (w << 4) + ln;
    const int bb   = blockIdx.x * 16;

    // byte strides inside the swizzled weight array
    const unsigned GSTRIDE = 16u * 9u * 32u * 32u;   // nt += 16  -> 147456 B
    const unsigned KSTRIDE = 32u * 32u;              // kc += 1   -> 1024 B

#pragma unroll
    for (int l = 0; l < 2; ++l) {
        float cst[8];
#pragma unroll
        for (int r = 0; r < 8; ++r) cst[r] = 0.0f;
        // h_{-1} = 0
#pragma unroll
        for (int r = 0; r < 8; ++r) hbuf[0][r + (lhi << 3)][k] = (__bf16)0.0f;
        // static part of aug region (both buffers): col 260 = 1, 261..295 = 0
        if (w == 14 && lane < 16) {
#pragma unroll
            for (int bufi = 0; bufi < 2; ++bufi) {
                hbuf[bufi][lane][260] = (__bf16)1.0f;
#pragma unroll
                for (int c = 261; c < 296; ++c) hbuf[bufi][lane][c] = (__bf16)0.0f;
            }
        }
        // stage x_0 into buf0 aug cols 256..259
        if (w == 15 && lane < 16) {
            const float* src = (l == 0) ? &x[((bb + lane) * TT) * 4]
                                        : &x1[(bb + lane) * 4];
            f32x4 xv = *(const f32x4*)src;
            hbuf[0][lane][256] = (__bf16)xv.x;
            hbuf[0][lane][257] = (__bf16)xv.y;
            hbuf[0][lane][258] = (__bf16)xv.z;
            hbuf[0][lane][259] = (__bf16)xv.w;
        }
        __syncthreads();

        const char* ubase = (const char*)(Ubs + (size_t)l * (64 * 9 * 32 * 16));
        const char* fbase = (const char*)(fws + (size_t)l * (9 * 32 * 16));

        for (int t = 0; t < TT; ++t) {
            const int cur = t & 1, nxt = cur ^ 1;
            // stage x_{t+1} into hbuf[nxt] aug
            if (w == 15 && lane < 16 && (t + 1) < TT) {
                const float* src = (l == 0) ? &x[((bb + lane) * TT + (t + 1)) * 4]
                                            : &x1[((t + 1) * BBATCH + bb + lane) * 4];
                f32x4 xv = *(const f32x4*)src;
                hbuf[nxt][lane][256] = (__bf16)xv.x;
                hbuf[nxt][lane][257] = (__bf16)xv.y;
                hbuf[nxt][lane][258] = (__bf16)xv.z;
                hbuf[nxt][lane][259] = (__bf16)xv.w;
            }

            // opaque per-lane byte offset: re-materialized every step so the 36
            // weight loads can't be hoisted out of the t-loop; pointer stays
            // typed -> global_load_b128 saddr+voffset+imm, no address VALU.
            unsigned voff = (unsigned)((w * 9 * 32 + lane) * 32);
            asm volatile("" : "+v"(voff));

            v8f acc[4] = {};
            FragBF A[2], Bm[2][4];
            // preload chunk 0
            {
                const int kb0 = (lhi << 3);
                A[0].u[0] = *(const u32x4*)&hbuf[cur][ln][kb0];
                A[0].u[1] = *(const u32x4*)&hbuf[cur][ln][kb0 + 16];
#pragma unroll
                for (int g = 0; g < 4; ++g) {
                    const u32x4* bp = (const u32x4*)(ubase + voff + g * GSTRIDE);
                    Bm[0][g].u[0] = bp[0];
                    Bm[0][g].u[1] = bp[1];
                }
            }
            // 2-stage pipeline over 9 K-chunks (256 h + 32 aug)
#pragma unroll
            for (int kc = 0; kc < 9; ++kc) {
                const int cb = kc & 1, nb = cb ^ 1;
                if (kc < 8) {
                    const int kb = (kc + 1) * 32 + (lhi << 3);
                    A[nb].u[0] = *(const u32x4*)&hbuf[cur][ln][kb];
                    A[nb].u[1] = *(const u32x4*)&hbuf[cur][ln][kb + 16];
#pragma unroll
                    for (int g = 0; g < 4; ++g) {
                        const u32x4* bp = (const u32x4*)(ubase + voff + g * GSTRIDE
                                                         + (kc + 1) * KSTRIDE);
                        Bm[nb][g].u[0] = bp[0];
                        Bm[nb][g].u[1] = bp[1];
                    }
                }
#pragma unroll
                for (int g = 0; g < 4; ++g)
                    acc[g] = __builtin_amdgcn_wmma_f32_16x16x32_bf16(
                        false, A[cb].v, false, Bm[cb][g].v, (short)0, acc[g], false, false);
            }

            // LSTM cell update; projections + biases already inside accumulators
            float hn[8];
#pragma unroll
            for (int r = 0; r < 8; ++r) {
                float ft = sigm(acc[0][r]);
                float it = sigm(acc[1][r]);
                float ot = sigm(acc[2][r]);
                float ct = tanh_fast(acc[3][r]);
                float c  = it * ct + ft * cst[r];
                cst[r]   = c;
                hn[r]    = ot * tanh_fast(c);
            }
#pragma unroll
            for (int r = 0; r < 8; ++r)
                hbuf[nxt][r + (lhi << 3)][k] = (__bf16)hn[r];

            // delayed fco tile (layer 0): out_{t-1} = h_{t-1}@fw + fb, normalized
            if (w == 0 && l == 0 && t > 0) {
                unsigned fvoff = (unsigned)(lane * 32);
                asm volatile("" : "+v"(fvoff));
                v8f afc = {};
#pragma unroll
                for (int kc = 0; kc < 9; ++kc) {
                    const int kb = kc * 32 + (lhi << 3);
                    FragBF Aa, Fm;
                    Aa.u[0] = *(const u32x4*)&hbuf[cur][ln][kb];
                    Aa.u[1] = *(const u32x4*)&hbuf[cur][ln][kb + 16];
                    const u32x4* fp = (const u32x4*)(fbase + fvoff + kc * KSTRIDE);
                    Fm.u[0] = fp[0];
                    Fm.u[1] = fp[1];
                    afc = __builtin_amdgcn_wmma_f32_16x16x32_bf16(
                        false, Aa.v, false, Fm.v, (short)0, afc, false, false);
                }
                fco_store(afc, &x1[(size_t)(t - 1) * BBATCH * 4], bb, lhi, ln);
            }
            __syncthreads();
        }

        // epilogue: out_{T-1} from h_{T-1} in hbuf[TT&1] (stale aug x rows hit
        // zero rows of the fco tile; constant-1 col still supplies fb)
        if (w == 0) {
            unsigned fvoff = (unsigned)(lane * 32);
            asm volatile("" : "+v"(fvoff));
            v8f afc = {};
#pragma unroll
            for (int kc = 0; kc < 9; ++kc) {
                const int kb = kc * 32 + (lhi << 3);
                FragBF Aa, Fm;
                Aa.u[0] = *(const u32x4*)&hbuf[TT & 1][ln][kb];
                Aa.u[1] = *(const u32x4*)&hbuf[TT & 1][ln][kb + 16];
                const u32x4* fp = (const u32x4*)(fbase + fvoff + kc * KSTRIDE);
                Fm.u[0] = fp[0];
                Fm.u[1] = fp[1];
                afc = __builtin_amdgcn_wmma_f32_16x16x32_bf16(
                    false, Aa.v, false, Fm.v, (short)0, afc, false, false);
            }
            float* dst = (l == 0) ? &x1[(size_t)(TT - 1) * BBATCH * 4] : out;
            fco_store(afc, dst, bb, lhi, ln);
        }
        __threadfence();   // x1 -> next layer's staging wave
        __syncthreads();
    }
}

// ---------------------------------------------------------------------------
extern "C" void kernel_launch(void* const* d_in, const int* in_sizes, int n_in,
                              void* d_out, int out_size, void* d_ws, size_t ws_size,
                              hipStream_t stream) {
    const float* x    = (const float*)d_in[0];
    const float* wxr  = (const float*)d_in[1];
    const float* wxi  = (const float*)d_in[2];
    const float* wxj  = (const float*)d_in[3];
    const float* wxk  = (const float*)d_in[4];
    const float* wxb  = (const float*)d_in[5];
    const float* uhr  = (const float*)d_in[6];
    const float* uhi  = (const float*)d_in[7];
    const float* uhj  = (const float*)d_in[8];
    const float* uhk  = (const float*)d_in[9];
    const float* fcow = (const float*)d_in[10];
    const float* fcob = (const float*)d_in[11];
    float* out = (float*)d_out;

    // workspace (~2.25 MB):
    char*   ws  = (char*)d_ws;
    float*  x1  = (float*)(ws);                 // [T][B][4] f32        = 1,048,576 B
    __bf16* Ubs = (__bf16*)(ws + 1048576);      // swizzled aug weights = 1,179,648 B
    __bf16* fws = (__bf16*)(ws + 2228224);      // swizzled fco tile    =    18,432 B

    prep_ubs<<<2304, 256, 0, stream>>>(uhr, uhi, uhj, uhk, wxr, wxi, wxj, wxk, wxb, Ubs);
    prep_fws<<<36, 256, 0, stream>>>(fcow, fcob, fws);
    qlstm_main<<<8, 512, 0, stream>>>(x, Ubs, fws, x1, out);
}
